// MoveNet_27290222199330
// MI455X (gfx1250) — matrix operand order repaired and checked by reading, not compiled, not run on hardware
//
#include <hip/hip_runtime.h>
#include <math.h>

typedef __attribute__((ext_vector_type(16))) _Float16     v16h;
typedef __attribute__((ext_vector_type(8)))  float        v8f;
typedef __attribute__((ext_vector_type(4)))  unsigned int u32x4;

union F16Frag { v16h h; u32x4 u[2]; };

#define FT    192
#define NPIX  (FT*FT)     // 36864
#define CF    24
#define HC    96

// ---------------------------------------------------------------------------
// Generic weight repack: src is (K x N) f32 row-major; dst is f16 B-fragment
// layout [kt][nt][lane][16] halves, K padded to KT*32, N padded to NT*16.
// Per ISA: lanes 0-15 hold K=kt*32+{0..7,16..23}, lanes 16-31 hold
// K=kt*32+{8..15,24..31}; column N = nt*16 + (lane&15).
// ---------------------------------------------------------------------------
__global__ void pack_w_kernel(const float* __restrict__ src, _Float16* __restrict__ dst,
                              int K, int N, int KT, int NT)
{
    int idx = blockIdx.x * 256 + threadIdx.x;
    int total = KT * NT * 512;
    if (idx >= total) return;
    int j    = idx & 15;
    int lane = (idx >> 4) & 31;
    int t    = idx >> 9;           // kt*NT + nt
    int nt   = t % NT;
    int kt   = t / NT;
    int cin  = kt * 32 + ((lane & 16) ? 8 : 0) + ((j & 8) ? 16 : 0) + (j & 7);
    int cout = nt * 16 + (lane & 15);
    float v  = (cin < K && cout < N) ? src[cin * N + cout] : 0.0f;
    dst[idx] = (_Float16)v;
}

// ---------------------------------------------------------------------------
// conv1: x(768,768,3) f32 -> f1(384,384,32) f16, 3x3 stride2 SAME, ReLU,
// input normalization fused. SAME pad for stride2/even: high-edge only.
// ---------------------------------------------------------------------------
__global__ void conv1_kernel(const float* __restrict__ x, const float* __restrict__ w,
                             const float* __restrict__ b, _Float16* __restrict__ f1)
{
    __shared__ float sw[864];
    __shared__ float sb[32];
    for (int i = threadIdx.x; i < 864; i += 256) sw[i] = w[i];
    if (threadIdx.x < 32) sb[threadIdx.x] = b[threadIdx.x];
    __syncthreads();

    int p = blockIdx.x * 256 + threadIdx.x;    // 0..147455
    if (p >= 384 * 384) return;
    int oy = p / 384, ox = p % 384;

    float acc[32];
    for (int c = 0; c < 32; ++c) acc[c] = sb[c];

    for (int ky = 0; ky < 3; ++ky) {
        int iy = 2 * oy + ky;
        if (iy >= 768) continue;
        for (int kx = 0; kx < 3; ++kx) {
            int ix = 2 * ox + kx;
            if (ix >= 768) continue;
            const float* px = x + (iy * 768 + ix) * 3;
            float x0 = px[0] * 0.007843137718737125f - 1.0f;
            float x1 = px[1] * 0.007843137718737125f - 1.0f;
            float x2 = px[2] * 0.007843137718737125f - 1.0f;
            const float* wp = sw + (ky * 3 + kx) * 96;  // (tap, cin, 32)
            for (int c = 0; c < 32; ++c)
                acc[c] += x0 * wp[c] + x1 * wp[32 + c] + x2 * wp[64 + c];
        }
    }
    _Float16* out = f1 + (size_t)p * 32;
    union { _Float16 h[8]; u32x4 u; } pk;
    for (int g = 0; g < 4; ++g) {
        for (int j = 0; j < 8; ++j) pk.h[j] = (_Float16)fmaxf(acc[g * 8 + j], 0.0f);
        *(u32x4*)(out + g * 8) = pk.u;
    }
}

// ---------------------------------------------------------------------------
// conv2 implicit-GEMM via WMMA: f1(384,384,32) f16 -> f(192,192,24) f16.
// 3x3 stride2 SAME (high-edge pad), +bias, ReLU.
// Wave = one 16-pixel M-tile; K = 9 taps x 32ch; N = 24 padded to 32 (2 tiles).
// ---------------------------------------------------------------------------
__global__ void conv2_wmma_kernel(const _Float16* __restrict__ f1,
                                  const _Float16* __restrict__ wpk,   // [9][2][32][16] halves
                                  const float* __restrict__ bias,
                                  _Float16* __restrict__ f)
{
    int wid  = threadIdx.x >> 5;
    int lane = threadIdx.x & 31;
    int tile = blockIdx.x * 8 + wid;          // 0..2303
    int m0   = tile * 16;
    int oy   = m0 / FT;                       // uniform in wave
    int ox   = (m0 % FT) + (lane & 15);
    int half = lane >> 4;
    int abase = half ? 8 : 0;
    u32x4 z4 = {};

    v8f acc0 = {}; v8f acc1 = {};

    for (int ky = 0; ky < 3; ++ky) {
        int iy = 2 * oy + ky;
        for (int kx = 0; kx < 3; ++kx) {
            int ix = 2 * ox + kx;
            bool valid = (iy < 384) && (ix < 384);
            int iyc = valid ? iy : 0;
            int ixc = valid ? ix : 0;
            const _Float16* ap = f1 + ((size_t)iyc * 384 + ixc) * 32 + abase;
            F16Frag a;
            a.u[0] = *(const u32x4*)(ap);        // K: base..base+7
            a.u[1] = *(const u32x4*)(ap + 16);   // K: base+16..base+23
            if (!valid) { a.u[0] = z4; a.u[1] = z4; }

            int tap = ky * 3 + kx;
            const _Float16* bp = wpk + ((size_t)(tap * 2) * 32 + lane) * 16;
            F16Frag b0, b1;
            b0.u[0] = *(const u32x4*)(bp);
            b0.u[1] = *(const u32x4*)(bp + 8);
            b1.u[0] = *(const u32x4*)(bp + 512);
            b1.u[1] = *(const u32x4*)(bp + 520);

            acc0 = __builtin_amdgcn_wmma_f32_16x16x32_f16(false, a.h, false, b0.h,
                                                          (short)0, acc0, false, false);
            acc1 = __builtin_amdgcn_wmma_f32_16x16x32_f16(false, a.h, false, b1.h,
                                                          (short)0, acc1, false, false);
        }
    }

    int col = lane & 15;
    for (int j = 0; j < 8; ++j) {
        int row = m0 + j + (half ? 8 : 0);
        float v0 = fmaxf(acc0[j] + bias[col], 0.0f);
        f[(size_t)row * CF + col] = (_Float16)v0;
        if (col < 8) {
            float v1 = fmaxf(acc1[j] + bias[16 + col], 0.0f);
            f[(size_t)row * CF + 16 + col] = (_Float16)v1;
        }
    }
}

// ---------------------------------------------------------------------------
// Depthwise 3x3 stride1 SAME (pad 1/1) per head: f(192,192,24) f16 -> h f16.
// No ReLU (reference has none after dw conv).
// ---------------------------------------------------------------------------
__global__ void dw_kernel(const _Float16* __restrict__ f, const float* __restrict__ w,
                          const float* __restrict__ b, _Float16* __restrict__ h)
{
    __shared__ float sw[216];
    __shared__ float sb[24];
    if (threadIdx.x < 216) sw[threadIdx.x] = w[threadIdx.x];
    if (threadIdx.x < 24)  sb[threadIdx.x] = b[threadIdx.x];
    __syncthreads();

    int p = blockIdx.x * 256 + threadIdx.x;
    if (p >= NPIX) return;
    int oy = p / FT, ox = p % FT;

    float acc[24];
    for (int c = 0; c < 24; ++c) acc[c] = sb[c];

    for (int ky = 0; ky < 3; ++ky) {
        int iy = oy + ky - 1;
        if (iy < 0 || iy >= FT) continue;
        for (int kx = 0; kx < 3; ++kx) {
            int ix = ox + kx - 1;
            if (ix < 0 || ix >= FT) continue;
            const _Float16* fp = f + ((size_t)iy * FT + ix) * CF;
            const float* wp = sw + (ky * 3 + kx) * 24;
            for (int c = 0; c < 24; ++c) acc[c] += (float)fp[c] * wp[c];
        }
    }
    _Float16* op = h + (size_t)p * CF;
    union { _Float16 hh[8]; u32x4 u; } pk;
    for (int g = 0; g < 3; ++g) {
        for (int j = 0; j < 8; ++j) pk.hh[j] = (_Float16)acc[g * 8 + j];
        *(u32x4*)(op + g * 8) = pk.u;
    }
}

// ---------------------------------------------------------------------------
// Fused head MLP via WMMA:
//   Y = relu(h @ W1 + b1)   (K=24 pad32, N=96: 6 tiles)
//   O = Y @ W2 + b2         (K=96: 3 tiles, N=C pad to NT2*16)
// Intermediate Y staged per-wave in LDS to convert C-layout -> A-layout.
// ---------------------------------------------------------------------------
__global__ void head_mlp_kernel(const _Float16* __restrict__ h,
                                const _Float16* __restrict__ pw1,  // [6][32][16]
                                const float* __restrict__ b1,
                                const _Float16* __restrict__ pw2,  // [3][NT2][32][16]
                                const float* __restrict__ b2,
                                float* __restrict__ out,
                                int C, int NT2, int dosig)
{
    __shared__ __align__(16) _Float16 Y[8][16 * HC];   // per-wave slice, 24 KB
    int wid  = threadIdx.x >> 5;
    int lane = threadIdx.x & 31;
    int tile = blockIdx.x * 8 + wid;
    int m0   = tile * 16;
    int half = lane >> 4;
    int col  = lane & 15;
    u32x4 z4 = {};

    // A fragment of h (row = m0+col; K chunk by half). Channels 24, pad to 32.
    F16Frag a1;
    {
        const _Float16* ap = h + ((size_t)(m0 + col)) * CF;
        if (half == 0) {
            a1.u[0] = *(const u32x4*)(ap);        // ch 0..7
            a1.u[1] = *(const u32x4*)(ap + 16);   // ch 16..23
        } else {
            a1.u[0] = *(const u32x4*)(ap + 8);    // ch 8..15
            a1.u[1] = z4;                         // ch 24..31 = pad
        }
    }

    v8f acc[6];
    for (int nt = 0; nt < 6; ++nt) acc[nt] = (v8f){};
    for (int nt = 0; nt < 6; ++nt) {
        const _Float16* bp = pw1 + ((size_t)(nt * 32 + lane)) * 16;
        F16Frag b;
        b.u[0] = *(const u32x4*)(bp);
        b.u[1] = *(const u32x4*)(bp + 8);
        acc[nt] = __builtin_amdgcn_wmma_f32_16x16x32_f16(false, a1.h, false, b.h,
                                                         (short)0, acc[nt], false, false);
    }

    // bias + relu -> LDS (f16, row-major 16x96)
    _Float16* y = &Y[wid][0];
    for (int nt = 0; nt < 6; ++nt) {
        float bb = b1[nt * 16 + col];
        for (int j = 0; j < 8; ++j) {
            int row = j + (half ? 8 : 0);
            float v = fmaxf(acc[nt][j] + bb, 0.0f);
            y[row * HC + nt * 16 + col] = (_Float16)v;
        }
    }
    // Same-wave cross-lane LDS dependence: DS ops are in-order, but force the
    // wait + a compiler barrier before re-reading other lanes' writes.
    asm volatile("s_wait_dscnt 0" ::: "memory");

    v8f acc2[3];
    for (int i = 0; i < 3; ++i) acc2[i] = (v8f){};
    int abase = half ? 8 : 0;
    for (int kt = 0; kt < 3; ++kt) {
        const _Float16* ap = y + col * HC + kt * 32;
        F16Frag a;
        a.u[0] = *(const u32x4*)(ap + abase);
        a.u[1] = *(const u32x4*)(ap + abase + 16);
        for (int nt = 0; nt < NT2; ++nt) {
            const _Float16* bp = pw2 + ((size_t)((kt * NT2 + nt) * 32 + lane)) * 16;
            F16Frag b;
            b.u[0] = *(const u32x4*)(bp);
            b.u[1] = *(const u32x4*)(bp + 8);
            acc2[nt] = __builtin_amdgcn_wmma_f32_16x16x32_f16(false, a.h, false, b.h,
                                                              (short)0, acc2[nt], false, false);
        }
    }

    for (int nt = 0; nt < NT2; ++nt) {
        int c = nt * 16 + col;
        if (c < C) {
            float bb = b2[c];
            for (int j = 0; j < 8; ++j) {
                int row = m0 + j + (half ? 8 : 0);
                float v = acc2[nt][j] + bb;
                if (dosig) v = 1.0f / (1.0f + __expf(-v));
                out[(size_t)row * C + c] = v;
            }
        }
    }
}

// ---------------------------------------------------------------------------
// Postprocess 1: argmax of sigmoid(hm)*w2c -> keypoint prior coords (34 f32).
// Tie-break = lowest linear index (matches jnp.argmax).
// ---------------------------------------------------------------------------
__global__ void post1_kernel(const float* __restrict__ hm_s, const float* __restrict__ hps,
                             float* __restrict__ scratch)
{
    __shared__ float sv[256];
    __shared__ int   si[256];
    float bv = -1e30f; int bi = 0;
    for (int p = threadIdx.x; p < NPIX; p += 256) {
        int yy = p / FT, xx = p % FT;
        float dy = (float)yy - 96.0f, dx = (float)xx - 96.0f;
        float w  = 1.0f / (sqrtf(dy * dy + dx * dx) + 1.8f);
        float v  = hm_s[p] * w;
        if (v > bv) { bv = v; bi = p; }
    }
    sv[threadIdx.x] = bv; si[threadIdx.x] = bi;
    __syncthreads();
    for (int s = 128; s > 0; s >>= 1) {
        if (threadIdx.x < s) {
            float ov = sv[threadIdx.x + s]; int oi = si[threadIdx.x + s];
            if (ov > sv[threadIdx.x] || (ov == sv[threadIdx.x] && oi < si[threadIdx.x])) {
                sv[threadIdx.x] = ov; si[threadIdx.x] = oi;
            }
        }
        __syncthreads();
    }
    if (threadIdx.x == 0) {
        int ct = si[0];
        float cty = floorf((float)ct / (float)FT);
        float ctx = (float)ct - cty * (float)FT;
        for (int k = 0; k < 17; ++k) {
            scratch[k * 2 + 0] = hps[(size_t)ct * 34 + k * 2 + 0] + cty;
            scratch[k * 2 + 1] = hps[(size_t)ct * 34 + k * 2 + 1] + ctx;
        }
    }
}

// ---------------------------------------------------------------------------
// Postprocess 2: per-keypoint distance-weighted argmax + gather -> 51 floats.
// ---------------------------------------------------------------------------
__global__ void post2_kernel(const float* __restrict__ hmhp_s, const float* __restrict__ hpoff,
                             const float* __restrict__ scratch, float* __restrict__ out)
{
    __shared__ float sv[256];
    __shared__ int   si[256];
    int k = blockIdx.x;
    float cy = scratch[k * 2 + 0];
    float cx = scratch[k * 2 + 1];
    float bv = -1e30f; int bi = 0;
    for (int p = threadIdx.x; p < NPIX; p += 256) {
        int yy = p / FT, xx = p % FT;
        float dy = (float)yy - cy, dx = (float)xx - cx;
        float s  = hmhp_s[(size_t)p * 17 + k] / (sqrtf(dy * dy + dx * dx) + 1.8f);
        if (s > bv) { bv = s; bi = p; }
    }
    sv[threadIdx.x] = bv; si[threadIdx.x] = bi;
    __syncthreads();
    for (int s = 128; s > 0; s >>= 1) {
        if (threadIdx.x < s) {
            float ov = sv[threadIdx.x + s]; int oi = si[threadIdx.x + s];
            if (ov > sv[threadIdx.x] || (ov == sv[threadIdx.x] && oi < si[threadIdx.x])) {
                sv[threadIdx.x] = ov; si[threadIdx.x] = oi;
            }
        }
        __syncthreads();
    }
    if (threadIdx.x == 0) {
        int top = si[0];
        float ky = floorf((float)top / (float)FT);
        float kx = (float)top - ky * (float)FT;
        float conf = hmhp_s[(size_t)top * 17 + k];
        out[k * 3 + 0] = (hpoff[(size_t)top * 34 + 2 * k + 0] + ky) / (float)FT;
        out[k * 3 + 1] = (hpoff[(size_t)top * 34 + 2 * k + 1] + kx) / (float)FT;
        out[k * 3 + 2] = conf;
    }
}

// ---------------------------------------------------------------------------
// Host launcher
// ---------------------------------------------------------------------------
extern "C" void kernel_launch(void* const* d_in, const int* in_sizes, int n_in,
                              void* d_out, int out_size, void* d_ws, size_t ws_size,
                              hipStream_t stream)
{
    const float* x      = (const float*)d_in[0];
    const float* bb_w1  = (const float*)d_in[1];
    const float* bb_b1  = (const float*)d_in[2];
    const float* bb_w2  = (const float*)d_in[3];
    const float* bb_b2  = (const float*)d_in[4];
    const float *dw_w[4], *dw_b[4], *w1[4], *b1[4], *w2[4], *b2[4];
    for (int hd = 0; hd < 4; ++hd) {
        int base = 5 + hd * 6;
        dw_w[hd] = (const float*)d_in[base + 0];
        dw_b[hd] = (const float*)d_in[base + 1];
        w1[hd]   = (const float*)d_in[base + 2];
        b1[hd]   = (const float*)d_in[base + 3];
        w2[hd]   = (const float*)d_in[base + 4];
        b2[hd]   = (const float*)d_in[base + 5];
    }
    const int Ccls[4] = {17, 1, 34, 34};
    const int NT2[4]  = {2, 1, 3, 3};
    const int SIG[4]  = {1, 1, 0, 0};  // hm_hp, hm sigmoided; hps, hp_offset raw

    // Workspace carve-up (256B aligned)
    char* ws = (char*)d_ws;
    size_t off = 0;
    auto carve = [&](size_t bytes) -> char* {
        char* p = ws + off;
        off = (off + bytes + 255) & ~(size_t)255;
        return p;
    };
    _Float16* f1   = (_Float16*)carve((size_t)384 * 384 * 32 * 2);  // conv1 out
    _Float16* f    = (_Float16*)carve((size_t)NPIX * CF * 2);       // conv2 out
    _Float16* hbuf[4];
    for (int hd = 0; hd < 4; ++hd) hbuf[hd] = (_Float16*)carve((size_t)NPIX * CF * 2);
    _Float16* pc2  = (_Float16*)carve((size_t)9 * 2 * 512 * 2);     // packed conv2 W
    _Float16* pw1[4];
    for (int hd = 0; hd < 4; ++hd) pw1[hd] = (_Float16*)carve((size_t)6 * 512 * 2);
    _Float16* pw2[4];
    for (int hd = 0; hd < 4; ++hd) pw2[hd] = (_Float16*)carve((size_t)3 * NT2[hd] * 512 * 2);
    float* outh[4];
    for (int hd = 0; hd < 4; ++hd) outh[hd] = (float*)carve((size_t)NPIX * Ccls[hd] * 4);
    float* scratch = (float*)carve(256);

    // --- weight repacks (f32 -> f16 B-fragment layout) ---
    pack_w_kernel<<<(9 * 2 * 512 + 255) / 256, 256, 0, stream>>>(bb_w2, pc2, 288, 24, 9, 2);
    for (int hd = 0; hd < 4; ++hd) {
        pack_w_kernel<<<(6 * 512 + 255) / 256, 256, 0, stream>>>(w1[hd], pw1[hd], 24, 96, 1, 6);
        pack_w_kernel<<<(3 * NT2[hd] * 512 + 255) / 256, 256, 0, stream>>>(
            w2[hd], pw2[hd], 96, Ccls[hd], 3, NT2[hd]);
    }

    // --- backbone ---
    conv1_kernel<<<(384 * 384) / 256, 256, 0, stream>>>(x, bb_w1, bb_b1, f1);
    conv2_wmma_kernel<<<(NPIX / 16) / 8, 256, 0, stream>>>(f1, pc2, bb_b2, f);

    // --- heads ---
    for (int hd = 0; hd < 4; ++hd)
        dw_kernel<<<NPIX / 256, 256, 0, stream>>>(f, dw_w[hd], dw_b[hd], hbuf[hd]);
    for (int hd = 0; hd < 4; ++hd)
        head_mlp_kernel<<<(NPIX / 16) / 8, 256, 0, stream>>>(
            hbuf[hd], pw1[hd], b1[hd], pw2[hd], b2[hd], outh[hd],
            Ccls[hd], NT2[hd], SIG[hd]);

    // --- postprocess ---
    post1_kernel<<<1, 256, 0, stream>>>(outh[1] /*hm*/, outh[2] /*hps*/, scratch);
    post2_kernel<<<17, 256, 0, stream>>>(outh[0] /*hm_hp*/, outh[3] /*hp_offset*/,
                                         scratch, (float*)d_out);
}
